// SetAbstraction_19121194402083
// MI455X (gfx1250) — matrix operand order repaired and checked by reading, compiled, not run on hardware
//
#include <hip/hip_runtime.h>
#include <hip/hip_bf16.h>
#include <math.h>

// ---------------------------------------------------------------------------
// Problem constants (from reference)
// ---------------------------------------------------------------------------
#define BATCH    8
#define NPER     4096
#define CIN      64
#define COUT     128
#define MDOWN    1024        // NPER * RATIO
#define KNN_K    32
#define DIN      67          // CIN + 3
#define LN_EPS   1e-5f
#define QPB      8           // queries processed per workgroup (amortize weights)

typedef __attribute__((ext_vector_type(16))) _Float16 v16h;
typedef __attribute__((ext_vector_type(8)))  float    v8f;

__device__ __forceinline__ float gelu_exact(float x) {
    return 0.5f * x * (1.0f + erff(x * 0.70710678118654752f));
}

// ---------------------------------------------------------------------------
// WMMA fragment builders (contiguous K order per lane half, wave32)
// ---------------------------------------------------------------------------
// A (16x32 f16): lanes 0-15 = rows 0-15 of the M-tile, K window base (lane>>4)*16;
// v16h elem i -> K = base+i.
__device__ __forceinline__ v16h load_afrag(const _Float16* sh, int stride,
                                           int mrow, int kb0, int lane) {
    int m  = mrow + (lane & 15);
    int kb = kb0 + ((lane >> 4) << 4);
    const _Float16* p = sh + m * stride + kb;
    v16h a;
#pragma unroll
    for (int i = 0; i < 8; ++i) a[i] = p[i];
#pragma unroll
    for (int i = 0; i < 8; ++i) a[8 + i] = p[8 + i];
    return a;
}

// B (32x16 f16, K x N): lane column n = n0 + (lane&15);
// lane K window = k0 + (lane>>4)*16; elem i -> K = window + i.
// Reads f32 weights (row-major [Kdim][COUT]) from global, converts, zero-pads K>=kvalid.
__device__ __forceinline__ v16h load_bfrag(const float* __restrict__ W, int ldw,
                                           int n0, int k0, int kvalid, int lane) {
    int n  = n0 + (lane & 15);
    int kb = k0 + ((lane >> 4) << 4);
    v16h b;
#pragma unroll
    for (int i = 0; i < 16; ++i) {
        int k   = kb + i;
        float v = (k < kvalid) ? W[(size_t)k * ldw + n] : 0.0f;
        b[i] = (_Float16)v;
    }
    return b;
}

// ---------------------------------------------------------------------------
// Kernel 1: exact KNN (top-32 smallest d2, stable tie-break by index).
// One wave per query; d2 cached in LDS; selection via strictly increasing
// 64-bit key = (f32 bits of d2 << 32) | (idx+1)  (monotone since d2 >= 0).
// ---------------------------------------------------------------------------
__global__ __launch_bounds__(128)
void knn_kernel(const float* __restrict__ pos, int* __restrict__ knn) {
    __shared__ float d2sh[4][NPER];                 // 64 KB
    const int tid  = threadIdx.x;
    const int lane = tid & 31;
    const int wave = tid >> 5;
    const int q = blockIdx.x * 4 + wave;            // 0 .. B*MDOWN-1
    const int b = q >> 10;                          // / MDOWN
    const int i = q & (MDOWN - 1);
    const float* posb = pos + (size_t)b * NPER * 3;
    const float cx = posb[i * 3 + 0];
    const float cy = posb[i * 3 + 1];
    const float cz = posb[i * 3 + 2];

    for (int j = lane; j < NPER; j += 32) {
        float dx = posb[j * 3 + 0] - cx;
        float dy = posb[j * 3 + 1] - cy;
        float dz = posb[j * 3 + 2] - cz;
        d2sh[wave][j] = dx * dx + dy * dy + dz * dz;
    }
    __syncthreads();

    unsigned long long prev = 0ull;
#pragma unroll 1
    for (int r = 0; r < KNN_K; ++r) {
        unsigned long long best = ~0ull;
        for (int j = lane; j < NPER; j += 32) {
            unsigned int bits = __float_as_uint(d2sh[wave][j]);
            unsigned long long key =
                ((unsigned long long)bits << 32) | (unsigned int)(j + 1);
            if (key > prev && key < best) best = key;
        }
#pragma unroll
        for (int off = 16; off; off >>= 1) {
            unsigned long long o = __shfl_xor(best, off, 32);
            if (o < best) best = o;
        }
        prev = best;
        if (lane == 0)
            knn[(size_t)q * KNN_K + r] = (int)(best & 0xFFFFFFFFu) - 1;
    }
}

// ---------------------------------------------------------------------------
// Kernel 2: fused gather -> GEMM1 -> LN+GeLU -> GEMM2 -> LN+GeLU -> col-max.
// One workgroup (8 waves) handles QPB queries. Wave w owns output columns
// [16w,16w+16). Weight B-fragments + LN params loaded ONCE per workgroup and
// held in VGPRs across all queries; intermediates stay in LDS.
// ---------------------------------------------------------------------------
__global__ __launch_bounds__(256)
void mlp_kernel(const float* __restrict__ x, const float* __restrict__ pos,
                const int* __restrict__ knn,
                const float* __restrict__ W1, const float* __restrict__ b1,
                const float* __restrict__ g1, const float* __restrict__ be1,
                const float* __restrict__ W2, const float* __restrict__ b2,
                const float* __restrict__ g2, const float* __restrict__ be2,
                float* __restrict__ out_x) {
    __shared__ _Float16 Ash[KNN_K][COUT];     // GEMM A input (f16), 8 KB
    __shared__ float    Ysh[KNN_K][COUT];     // GEMM D / activations (f32), 16 KB
    __shared__ int      idxsh[KNN_K];

    const int tid  = threadIdx.x;
    const int lane = tid & 31;
    const int wave = tid >> 5;                // 0..7
    const int n0   = wave * 16;               // this wave's N-slice
    const int qbase = blockIdx.x * QPB;

    if (tid == 0) { __builtin_prefetch(W1, 0, 0); __builtin_prefetch(W2, 0, 0); }

    // ---- per-lane LN parameters (columns lane*4 .. lane*4+3), loaded once
    float b1v[4], g1v[4], be1v[4], b2v[4], g2v[4], be2v[4];
#pragma unroll
    for (int u = 0; u < 4; ++u) {
        int c = lane * 4 + u;
        b1v[u] = b1[c];  g1v[u] = g1[c];  be1v[u] = be1[c];
        b2v[u] = b2[c];  g2v[u] = g2[c];  be2v[u] = be2[c];
    }

    // ---- per-wave weight fragments (held in VGPRs for the whole kernel)
    v16h B1f[3], B2f[4];
#pragma unroll
    for (int s = 0; s < 3; ++s) B1f[s] = load_bfrag(W1, COUT, n0, s * 32, DIN, lane);
#pragma unroll
    for (int s = 0; s < 4; ++s) B2f[s] = load_bfrag(W2, COUT, n0, s * 32, COUT, lane);

#pragma unroll 1
    for (int qi = 0; qi < QPB; ++qi) {
        const int q = qbase + qi;             // 0 .. B*MDOWN-1
        const int b = q >> 10;
        const int i = q & (MDOWN - 1);

        if (tid < KNN_K) idxsh[tid] = knn[(size_t)q * KNN_K + tid];
        __syncthreads();

        const float* posb = pos + (size_t)b * NPER * 3;
        const float cx = posb[i * 3 + 0];
        const float cy = posb[i * 3 + 1];
        const float cz = posb[i * 3 + 2];

        // ---- gather: rows = 32 neighbors; cols 0..63 feat, 64..66 rel pos, rest 0
        for (int e = tid; e < KNN_K * COUT; e += 256) {
            int r = e >> 7;
            int c = e & (COUT - 1);
            int j = idxsh[r];
            float v;
            if (c < CIN)        v = x[((size_t)b * NPER + j) * CIN + c];
            else if (c == 64)   v = posb[j * 3 + 0] - cx;
            else if (c == 65)   v = posb[j * 3 + 1] - cy;
            else if (c == 66)   v = posb[j * 3 + 2] - cz;
            else                v = 0.0f;
            Ash[r][c] = (_Float16)v;
        }
        __syncthreads();

        // ---- GEMM1: (32 x 96) @ (96 x 128), 2 M-tiles x 3 K-steps per wave
        v8f c0 = {}; v8f c1 = {};
#pragma unroll
        for (int s = 0; s < 3; ++s) {
            v16h a0 = load_afrag(&Ash[0][0], COUT, 0,  s * 32, lane);
            v16h a1 = load_afrag(&Ash[0][0], COUT, 16, s * 32, lane);
            c0 = __builtin_amdgcn_wmma_f32_16x16x32_f16(false, a0, false, B1f[s],
                                                        (short)0, c0, false, false);
            c1 = __builtin_amdgcn_wmma_f32_16x16x32_f16(false, a1, false, B1f[s],
                                                        (short)0, c1, false, false);
        }
        // D layout: VGPR r -> row (lane>>4)*8 + r, col n0 + (lane&15)
        {
            int colw  = n0 + (lane & 15);
            int rbase = (lane >> 4) << 3;
#pragma unroll
            for (int r = 0; r < 8; ++r) {
                Ysh[rbase + r][colw]      = c0[r];
                Ysh[16 + rbase + r][colw] = c1[r];
            }
        }
        __syncthreads();

        // ---- LN1 + exact GeLU: wave w handles rows 4w..4w+3; write f16 A for GEMM2
#pragma unroll
        for (int rr = 0; rr < 4; ++rr) {
            int r = wave * 4 + rr;
            float v[4], s = 0.f, ss = 0.f;
#pragma unroll
            for (int u = 0; u < 4; ++u) {
                float t = Ysh[r][lane * 4 + u] + b1v[u];
                v[u] = t; s += t; ss += t * t;
            }
#pragma unroll
            for (int off = 16; off; off >>= 1) {
                s  += __shfl_xor(s,  off, 32);
                ss += __shfl_xor(ss, off, 32);
            }
            float mu   = s * (1.0f / COUT);
            float var  = ss * (1.0f / COUT) - mu * mu;
            float rstd = rsqrtf(var + LN_EPS);
#pragma unroll
            for (int u = 0; u < 4; ++u) {
                float t = (v[u] - mu) * rstd * g1v[u] + be1v[u];
                Ash[r][lane * 4 + u] = (_Float16)gelu_exact(t);
            }
        }
        __syncthreads();

        // ---- GEMM2: (32 x 128) @ (128 x 128), 4 K-steps
        v8f d0 = {}; v8f d1 = {};
#pragma unroll
        for (int s = 0; s < 4; ++s) {
            v16h a0 = load_afrag(&Ash[0][0], COUT, 0,  s * 32, lane);
            v16h a1 = load_afrag(&Ash[0][0], COUT, 16, s * 32, lane);
            d0 = __builtin_amdgcn_wmma_f32_16x16x32_f16(false, a0, false, B2f[s],
                                                        (short)0, d0, false, false);
            d1 = __builtin_amdgcn_wmma_f32_16x16x32_f16(false, a1, false, B2f[s],
                                                        (short)0, d1, false, false);
        }
        {
            int colw  = n0 + (lane & 15);
            int rbase = (lane >> 4) << 3;
#pragma unroll
            for (int r = 0; r < 8; ++r) {
                Ysh[rbase + r][colw]      = d0[r];
                Ysh[16 + rbase + r][colw] = d1[r];
            }
        }
        __syncthreads();

        // ---- LN2 + exact GeLU (result back into Ysh as f32)
#pragma unroll
        for (int rr = 0; rr < 4; ++rr) {
            int r = wave * 4 + rr;
            float v[4], s = 0.f, ss = 0.f;
#pragma unroll
            for (int u = 0; u < 4; ++u) {
                float t = Ysh[r][lane * 4 + u] + b2v[u];
                v[u] = t; s += t; ss += t * t;
            }
#pragma unroll
            for (int off = 16; off; off >>= 1) {
                s  += __shfl_xor(s,  off, 32);
                ss += __shfl_xor(ss, off, 32);
            }
            float mu   = s * (1.0f / COUT);
            float var  = ss * (1.0f / COUT) - mu * mu;
            float rstd = rsqrtf(var + LN_EPS);
#pragma unroll
            for (int u = 0; u < 4; ++u) {
                float t = (v[u] - mu) * rstd * g2v[u] + be2v[u];
                Ysh[r][lane * 4 + u] = gelu_exact(t);
            }
        }
        __syncthreads();

        // ---- column max over the 32 neighbors
        if (tid < COUT) {
            float mx = Ysh[0][tid];
#pragma unroll
            for (int r = 1; r < KNN_K; ++r) mx = fmaxf(mx, Ysh[r][tid]);
            out_x[(size_t)q * COUT + tid] = mx;
        }
        // next iteration's post-gather barrier orders Ysh/Ash reuse vs. these reads
    }
}

// ---------------------------------------------------------------------------
// Kernel 3: pos_down + batch_down (trivial copy)
// ---------------------------------------------------------------------------
__global__ __launch_bounds__(256)
void posdown_kernel(const float* __restrict__ pos,
                    float* __restrict__ out_pos, float* __restrict__ out_batch) {
    int t = blockIdx.x * 256 + threadIdx.x;     // 0 .. B*MDOWN-1
    if (t >= BATCH * MDOWN) return;
    int b = t >> 10;
    int i = t & (MDOWN - 1);
    const float* p = pos + ((size_t)b * NPER + i) * 3;
    out_pos[t * 3 + 0] = p[0];
    out_pos[t * 3 + 1] = p[1];
    out_pos[t * 3 + 2] = p[2];
    out_batch[t] = (float)b;
}

// ---------------------------------------------------------------------------
// Kernel 4: upsample_idx — nearest downsampled point per original point
// (argmin with first-occurrence tie-break via lexicographic 64-bit key).
// One wave per point.
// ---------------------------------------------------------------------------
__global__ __launch_bounds__(256)
void upsample_kernel(const float* __restrict__ pos, float* __restrict__ out_up) {
    const int tid  = threadIdx.x;
    const int lane = tid & 31;
    const int wave = tid >> 5;
    const int p = blockIdx.x * 8 + wave;        // 0 .. B*NPER-1
    const int b = p >> 12;
    const int j = p & (NPER - 1);
    const float* posb = pos + (size_t)b * NPER * 3;
    const float cx = posb[j * 3 + 0];
    const float cy = posb[j * 3 + 1];
    const float cz = posb[j * 3 + 2];

    unsigned long long best = ~0ull;
    for (int i = lane; i < MDOWN; i += 32) {
        float dx = posb[i * 3 + 0] - cx;
        float dy = posb[i * 3 + 1] - cy;
        float dz = posb[i * 3 + 2] - cz;
        float d2 = dx * dx + dy * dy + dz * dz;
        unsigned long long key =
            ((unsigned long long)__float_as_uint(d2) << 32) | (unsigned int)i;
        if (key < best) best = key;
    }
#pragma unroll
    for (int off = 16; off; off >>= 1) {
        unsigned long long o = __shfl_xor(best, off, 32);
        if (o < best) best = o;
    }
    if (lane == 0)
        out_up[p] = (float)((int)(best & 0xFFFFFFFFu) + b * MDOWN);
}

// ---------------------------------------------------------------------------
// Host launcher
// ---------------------------------------------------------------------------
extern "C" void kernel_launch(void* const* d_in, const int* in_sizes, int n_in,
                              void* d_out, int out_size, void* d_ws, size_t ws_size,
                              hipStream_t stream) {
    const float* x   = (const float*)d_in[0];
    const float* pos = (const float*)d_in[1];
    // d_in[2] = batch (unused: implied by layout)
    const float* W1  = (const float*)d_in[3];
    const float* b1  = (const float*)d_in[4];
    const float* g1  = (const float*)d_in[5];
    const float* be1 = (const float*)d_in[6];
    const float* W2  = (const float*)d_in[7];
    const float* b2  = (const float*)d_in[8];
    const float* g2  = (const float*)d_in[9];
    const float* be2 = (const float*)d_in[10];

    float* out = (float*)d_out;
    float* out_xdown = out;                                       // 8192*128
    float* out_pos   = out_xdown + (size_t)BATCH * MDOWN * COUT;  // 8192*3
    float* out_batch = out_pos + (size_t)BATCH * MDOWN * 3;       // 8192
    float* out_up    = out_batch + (size_t)BATCH * MDOWN;         // 32768

    int* knn = (int*)d_ws;                                        // 8192*32 ints

    // 1) KNN: 8192 queries, 4 waves (queries) per 128-thread block
    knn_kernel<<<(BATCH * MDOWN) / 4, 128, 0, stream>>>(pos, knn);

    // 2) fused MLP: one 256-thread block per QPB queries
    mlp_kernel<<<(BATCH * MDOWN) / QPB, 256, 0, stream>>>(x, pos, knn,
                                                          W1, b1, g1, be1,
                                                          W2, b2, g2, be2,
                                                          out_xdown);

    // 3) pos_down / batch_down
    posdown_kernel<<<(BATCH * MDOWN + 255) / 256, 256, 0, stream>>>(
        pos, out_pos, out_batch);

    // 4) upsample argmin: 8 waves (points) per 256-thread block
    upsample_kernel<<<(BATCH * NPER) / 8, 256, 0, stream>>>(pos, out_up);
}
